// MHA_21741124452634
// MI455X (gfx1250) — compile-verified, matrix operand order checked by reading, not run on hardware
//
#include <hip/hip_runtime.h>
#include <hip/hip_bf16.h>

// MHA forward for MI455X (gfx1250): bf16 WMMA GEMMs + async-LDS flash attention.
// x:[4,2048,1024] f32 -> out:[4,2048,1024] f32
// ws layout (bf16): Q[64bh][2048][64] | K[64bh][2048][64] | Vt[64bh][64][2048] | AO[8192][1024]

typedef __attribute__((ext_vector_type(16))) __bf16 bf16x16;
typedef __attribute__((ext_vector_type(8)))  __bf16 bf16x8;
typedef __attribute__((ext_vector_type(4)))  __bf16 bf16x4;
typedef __attribute__((ext_vector_type(8)))  float  f32x8;
typedef __attribute__((ext_vector_type(4)))  float  f32x4;

#define DM   1024
#define SEQ  2048
#define NH   16
#define DH   64

__device__ __forceinline__ f32x8 wmma_bf16(bf16x16 a, bf16x16 b, f32x8 c) {
  // D = A(16x32 bf16) * B(32x16 bf16) + C(16x16 f32)
  return __builtin_amdgcn_wmma_f32_16x16x32_bf16(false, a, false, b, (short)0, c, false, false);
}

// Async DMA: 16B per lane, global -> LDS, tracked by ASYNCcnt (CDNA5).
__device__ __forceinline__ void async_b128(unsigned int lds_off, const void* gptr) {
  asm volatile("global_load_async_to_lds_b128 %0, %1, off"
               :: "v"(lds_off), "v"(gptr) : "memory");
}
__device__ __forceinline__ void wait_async0() {
  asm volatile("s_wait_asynccnt 0" ::: "memory");
}
__device__ __forceinline__ void wait_ds0() {
  asm volatile("s_wait_dscnt 0" ::: "memory");
}
__device__ __forceinline__ unsigned int lds_addr(const void* p) {
  return (unsigned int)(uintptr_t)p;   // low 32 bits of a shared generic ptr = LDS offset
}

// Load a 16x32 bf16 operand fragment from an LDS tile (rows along the lane-held
// index, 32 contiguous K elements per row). Layout per ISA 7.12.2:
// lanes 0-15 hold K 0..7,16..23 ; lanes 16-31 hold K 8..15,24..31.
__device__ __forceinline__ bf16x16 lds_frag(const __bf16* base, int stride, int lane) {
  const __bf16* p = base + (lane & 15) * stride + ((lane >> 4) << 3);
  bf16x8 lo = *(const bf16x8*)(p);
  bf16x8 hi = *(const bf16x8*)(p + 16);
  return __builtin_shufflevector(lo, hi, 0,1,2,3,4,5,6,7,8,9,10,11,12,13,14,15);
}

// ---------------------------------------------------------------------------
// Kernel 1: QKV projection.  out = x @ W^T + b, cast bf16, attention layouts.
// Register-staged double-buffered LDS: one barrier / k-step, global loads for
// step k+1 in flight across the WMMA block of step k.
// grid: (8192/128, 1024/128, 3)   block: 256 (8 waves)
// ---------------------------------------------------------------------------
__global__ __launch_bounds__(256)
void qkv_gemm_kernel(const float* __restrict__ x,
                     const float* __restrict__ Wq, const float* __restrict__ bq,
                     const float* __restrict__ Wk, const float* __restrict__ bk,
                     const float* __restrict__ Wv, const float* __restrict__ bv,
                     __bf16* __restrict__ Q, __bf16* __restrict__ K,
                     __bf16* __restrict__ Vt)
{
  constexpr int LDA = 40;                       // 80B row stride: 16B aligned, staggered banks
  __shared__ __bf16 sA[2][128 * LDA];
  __shared__ __bf16 sB[2][128 * LDA];

  const int which = blockIdx.z;
  const float* __restrict__ W    = (which == 0) ? Wq : (which == 1) ? Wk : Wv;
  const float* __restrict__ bias = (which == 0) ? bq : (which == 1) ? bk : bv;

  const int tid   = threadIdx.x;
  const int lane  = tid & 31;
  const int wave  = tid >> 5;
  const int waveM = (wave & 1) * 64;            // 2 waves along M
  const int waveN = (wave >> 1) * 32;           // 4 waves along N
  const int rowBase = blockIdx.x * 128;
  const int colBase = blockIdx.y * 128;

  const f32x8 zero8 = {0.f,0.f,0.f,0.f,0.f,0.f,0.f,0.f};
  f32x8 acc[4][2];
  #pragma unroll
  for (int mi = 0; mi < 4; ++mi)
    #pragma unroll
    for (int ni = 0; ni < 2; ++ni) acc[mi][ni] = zero8;

  f32x4 ra[4], rbv[4];                          // register staging for next tile
  auto gload = [&](int kt) {
    #pragma unroll
    for (int i = 0; i < 4; ++i) {
      int q = tid + i * 256; int r = q >> 3; int c4 = (q & 7) << 2;
      ra[i]  = *(const f32x4*)(x + (size_t)(rowBase + r) * DM + kt * 32 + c4);
      rbv[i] = *(const f32x4*)(W + (size_t)(colBase + r) * DM + kt * 32 + c4);
    }
  };
  auto lstore = [&](int buf) {
    #pragma unroll
    for (int i = 0; i < 4; ++i) {
      int q = tid + i * 256; int r = q >> 3; int c4 = (q & 7) << 2;
      bf16x4 ha, hb;
      #pragma unroll
      for (int j = 0; j < 4; ++j) { ha[j] = (__bf16)ra[i][j]; hb[j] = (__bf16)rbv[i][j]; }
      *(bf16x4*)(&sA[buf][r * LDA + c4]) = ha;
      *(bf16x4*)(&sB[buf][r * LDA + c4]) = hb;
    }
  };

  gload(0);
  lstore(0);
  __syncthreads();

  for (int kt = 0; kt < DM / 32; ++kt) {
    const int cur = kt & 1;
    if (kt + 1 < DM / 32) gload(kt + 1);        // in flight across WMMAs below
    if (kt + 2 < DM / 32) {
      __builtin_prefetch(x + (size_t)(rowBase + (tid >> 1)) * DM + (kt + 2) * 32, 0, 1);
      __builtin_prefetch(W + (size_t)(colBase + (tid >> 1)) * DM + (kt + 2) * 32, 0, 1);
    }

    bf16x16 af[4], bfr[2];
    #pragma unroll
    for (int mi = 0; mi < 4; ++mi) af[mi]  = lds_frag(&sA[cur][(waveM + mi * 16) * LDA], LDA, lane);
    #pragma unroll
    for (int ni = 0; ni < 2; ++ni) bfr[ni] = lds_frag(&sB[cur][(waveN + ni * 16) * LDA], LDA, lane);
    #pragma unroll
    for (int mi = 0; mi < 4; ++mi)
      #pragma unroll
      for (int ni = 0; ni < 2; ++ni)
        acc[mi][ni] = wmma_bf16(af[mi], bfr[ni], acc[mi][ni]);

    if (kt + 1 < DM / 32) {
      lstore(cur ^ 1);                          // convert + park next tile
      __syncthreads();                          // single barrier per k-step
    }
  }

  // Epilogue: bias, Q-scale (1/sqrt(64) * log2(e) folded for exp2 softmax), layout stores.
  const int cn = lane & 15;
  const int rb = (lane >> 4) * 8;               // C layout: vgpr i = row rb+i, col cn
  const float qscale = 0.125f * 1.44269504088896f;
  #pragma unroll
  for (int mi = 0; mi < 4; ++mi) {
    #pragma unroll
    for (int ni = 0; ni < 2; ++ni) {
      const int gn = colBase + waveN + ni * 16 + cn;
      const float bv_ = bias[gn];
      const int h = gn >> 6, d = gn & 63;
      #pragma unroll
      for (int i = 0; i < 8; ++i) {
        const int gm = rowBase + waveM + mi * 16 + rb + i;
        const int bidx = gm >> 11, s = gm & (SEQ - 1);
        const size_t bh = (size_t)(bidx * NH + h);
        const float v = acc[mi][ni][i] + bv_;
        if (which == 0)      Q [(bh * SEQ + s) * DH + d] = (__bf16)(v * qscale);
        else if (which == 1) K [(bh * SEQ + s) * DH + d] = (__bf16)v;
        else                 Vt[(bh * DH + d) * SEQ + s] = (__bf16)v;
      }
    }
  }
}

// ---------------------------------------------------------------------------
// Kernel 2: flash attention.  grid: (2048/128, 16, 4)  block: 256 (8 waves).
// Double-buffered K/V tiles filled by GLOBAL_LOAD_ASYNC_TO_LDS_B128 (ASYNCcnt),
// one barrier per kv-tile; Q tile region reused as per-wave P scratch.
// ---------------------------------------------------------------------------
__global__ __launch_bounds__(256)
void flash_attn_kernel(const __bf16* __restrict__ Q, const __bf16* __restrict__ K,
                       const __bf16* __restrict__ Vt, __bf16* __restrict__ AO)
{
  constexpr int LD = 72;                        // 144B row stride (16B aligned)
  __shared__ __bf16 smem[128 * LD + 4 * 64 * LD];   // 55,296 B
  __bf16* sQP = smem;                           // 128x72: Q tile, then P scratch
  __bf16* sK0 = smem + 128 * LD;
  __bf16* sK1 = sK0 + 64 * LD;
  __bf16* sV0 = sK1 + 64 * LD;                  // Vt tile: row = d, cols = s
  __bf16* sV1 = sV0 + 64 * LD;

  const int tid = threadIdx.x, lane = tid & 31, wave = tid >> 5;
  const int qt = blockIdx.x, h = blockIdx.y, b = blockIdx.z;
  const size_t bh = (size_t)(b * NH + h);
  const __bf16* __restrict__ Qp = Q  + bh * SEQ * DH;
  const __bf16* __restrict__ Kp = K  + bh * SEQ * DH;
  const __bf16* __restrict__ Vp = Vt + bh * DH * SEQ;

  auto issue_kv = [&](int kv, __bf16* kb, __bf16* vb) {
    #pragma unroll
    for (int i = 0; i < 2; ++i) {               // 64x64 bf16 per tile, 16B per lane
      int q = tid + i * 256; int r = q >> 3; int c = (q & 7) << 3;
      async_b128(lds_addr(kb + r * LD + c), Kp + (size_t)(kv * 64 + r) * DH + c);
      async_b128(lds_addr(vb + r * LD + c), Vp + (size_t)r * SEQ + (size_t)kv * 64 + c);
    }
  };

  // Prologue: async-load Q tile (128x64) + first K/V tiles, then wait+barrier.
  #pragma unroll
  for (int i = 0; i < 4; ++i) {
    int q = tid + i * 256; int r = q >> 3; int c = (q & 7) << 3;
    async_b128(lds_addr(sQP + r * LD + c), Qp + (size_t)(qt * 128 + r) * DH + c);
  }
  issue_kv(0, sK0, sV0);
  wait_async0();
  __syncthreads();

  const bf16x16 qf0 = lds_frag(sQP + (wave * 16) * LD + 0,  LD, lane);
  const bf16x16 qf1 = lds_frag(sQP + (wave * 16) * LD + 32, LD, lane);
  __bf16* sP = sQP + (wave * 16) * LD;          // reuse own Q rows as P scratch

  const f32x8 zero8 = {0.f,0.f,0.f,0.f,0.f,0.f,0.f,0.f};
  f32x8 o[4];
  float m[8], l[8];
  #pragma unroll
  for (int ds = 0; ds < 4; ++ds) o[ds] = zero8;
  #pragma unroll
  for (int i = 0; i < 8; ++i) { m[i] = -3.0e38f; l[i] = 0.f; }

  const int cn = lane & 15;
  const int rb = (lane >> 4) * 8;

  for (int kv = 0; kv < SEQ / 64; ++kv) {
    __bf16* cK = (kv & 1) ? sK1 : sK0;
    __bf16* cV = (kv & 1) ? sV1 : sV0;
    if (kv) {                                   // current tile landed? (own ops + all waves)
      wait_async0();
      __syncthreads();
    }
    if (kv + 1 < SEQ / 64)                      // DMA next tile across this iteration
      issue_kv(kv + 1, (kv & 1) ? sK0 : sK1, (kv & 1) ? sV0 : sV1);

    // S(16x64) = Q(16x64) @ K^T : B-operand rows = kv columns, elements along d
    f32x8 s[4];
    #pragma unroll
    for (int ni = 0; ni < 4; ++ni) {
      s[ni] = zero8;
      bf16x16 kf0 = lds_frag(cK + (ni * 16) * LD + 0,  LD, lane);
      bf16x16 kf1 = lds_frag(cK + (ni * 16) * LD + 32, LD, lane);
      s[ni] = wmma_bf16(qf0, kf0, s[ni]);
      s[ni] = wmma_bf16(qf1, kf1, s[ni]);
    }

    // online softmax: row reductions live in 16-lane halves (C layout)
    float corr[8], rs[8];
    #pragma unroll
    for (int i = 0; i < 8; ++i) {
      float v = fmaxf(fmaxf(s[0][i], s[1][i]), fmaxf(s[2][i], s[3][i]));
      v = fmaxf(v, __shfl_xor(v, 1, 16));
      v = fmaxf(v, __shfl_xor(v, 2, 16));
      v = fmaxf(v, __shfl_xor(v, 4, 16));
      v = fmaxf(v, __shfl_xor(v, 8, 16));
      const float mn = fmaxf(m[i], v);
      corr[i] = exp2f(m[i] - mn);
      m[i] = mn;
      rs[i] = 0.f;
    }
    #pragma unroll
    for (int ni = 0; ni < 4; ++ni)
      #pragma unroll
      for (int i = 0; i < 8; ++i) {
        const float p = exp2f(s[ni][i] - m[i]);
        rs[i] += p;
        sP[(rb + i) * LD + ni * 16 + cn] = (__bf16)p;   // C layout -> LDS
      }
    #pragma unroll
    for (int i = 0; i < 8; ++i) {
      float t = rs[i];
      t += __shfl_xor(t, 1, 16);
      t += __shfl_xor(t, 2, 16);
      t += __shfl_xor(t, 4, 16);
      t += __shfl_xor(t, 8, 16);
      l[i] = l[i] * corr[i] + t;
    }
    #pragma unroll
    for (int ds = 0; ds < 4; ++ds)
      #pragma unroll
      for (int i = 0; i < 8; ++i) o[ds][i] *= corr[i];

    // same-wave LDS RAW: drain DS queue before reading P back in A layout
    wait_ds0();
    const bf16x16 pf0 = lds_frag(sP + 0,  LD, lane);
    const bf16x16 pf1 = lds_frag(sP + 32, LD, lane);
    #pragma unroll
    for (int ds = 0; ds < 4; ++ds) {            // O += P(16x64) @ V(64x64-slice)
      bf16x16 vf0 = lds_frag(cV + (ds * 16) * LD + 0,  LD, lane);
      bf16x16 vf1 = lds_frag(cV + (ds * 16) * LD + 32, LD, lane);
      o[ds] = wmma_bf16(pf0, vf0, o[ds]);
      o[ds] = wmma_bf16(pf1, vf1, o[ds]);
    }
  }

  #pragma unroll
  for (int i = 0; i < 8; ++i) l[i] = 1.0f / l[i];
  #pragma unroll
  for (int ds = 0; ds < 4; ++ds)
    #pragma unroll
    for (int i = 0; i < 8; ++i) {
      const int srow = qt * 128 + wave * 16 + rb + i;
      const int col  = h * DH + ds * 16 + cn;
      AO[(size_t)(b * SEQ + srow) * DM + col] = (__bf16)(o[ds][i] * l[i]);
    }
}

// ---------------------------------------------------------------------------
// Kernel 3: output projection.  out = AO(bf16) @ Wo^T + bo  (f32 out)
// A operand: async-to-LDS double buffer.  B operand: reg-staged f32->bf16.
// grid: (8192/128, 1024/128)   block: 256
// ---------------------------------------------------------------------------
__global__ __launch_bounds__(256)
void out_gemm_kernel(const __bf16* __restrict__ A, const float* __restrict__ Wo,
                     const float* __restrict__ bo, float* __restrict__ out)
{
  constexpr int LDA = 40;
  __shared__ __bf16 sA[2][128 * LDA];
  __shared__ __bf16 sB[2][128 * LDA];

  const int tid   = threadIdx.x;
  const int lane  = tid & 31;
  const int wave  = tid >> 5;
  const int waveM = (wave & 1) * 64;
  const int waveN = (wave >> 1) * 32;
  const int rowBase = blockIdx.x * 128;
  const int colBase = blockIdx.y * 128;

  auto issueA = [&](int kt, int buf) {          // 128x32 bf16 via async DMA
    #pragma unroll
    for (int i = 0; i < 2; ++i) {
      int q = tid + i * 256; int r = q >> 2; int c = (q & 3) << 3;
      async_b128(lds_addr(&sA[buf][r * LDA + c]),
                 A + (size_t)(rowBase + r) * DM + kt * 32 + c);
    }
  };
  f32x4 rbv[4];
  auto gloadB = [&](int kt) {
    #pragma unroll
    for (int i = 0; i < 4; ++i) {
      int q = tid + i * 256; int r = q >> 3; int c4 = (q & 7) << 2;
      rbv[i] = *(const f32x4*)(Wo + (size_t)(colBase + r) * DM + kt * 32 + c4);
    }
  };
  auto lstoreB = [&](int buf) {
    #pragma unroll
    for (int i = 0; i < 4; ++i) {
      int q = tid + i * 256; int r = q >> 3; int c4 = (q & 7) << 2;
      bf16x4 hb;
      #pragma unroll
      for (int j = 0; j < 4; ++j) hb[j] = (__bf16)rbv[i][j];
      *(bf16x4*)(&sB[buf][r * LDA + c4]) = hb;
    }
  };

  const f32x8 zero8 = {0.f,0.f,0.f,0.f,0.f,0.f,0.f,0.f};
  f32x8 acc[4][2];
  #pragma unroll
  for (int mi = 0; mi < 4; ++mi)
    #pragma unroll
    for (int ni = 0; ni < 2; ++ni) acc[mi][ni] = zero8;

  issueA(0, 0);
  gloadB(0);
  lstoreB(0);
  wait_async0();
  __syncthreads();

  for (int kt = 0; kt < DM / 32; ++kt) {
    const int cur = kt & 1;
    if (kt + 1 < DM / 32) {
      issueA(kt + 1, cur ^ 1);                  // async DMA across WMMAs below
      gloadB(kt + 1);
    }
    if (kt + 2 < DM / 32)
      __builtin_prefetch(Wo + (size_t)(colBase + (tid >> 1)) * DM + (kt + 2) * 32, 0, 1);

    bf16x16 af[4], bfr[2];
    #pragma unroll
    for (int mi = 0; mi < 4; ++mi) af[mi]  = lds_frag(&sA[cur][(waveM + mi * 16) * LDA], LDA, lane);
    #pragma unroll
    for (int ni = 0; ni < 2; ++ni) bfr[ni] = lds_frag(&sB[cur][(waveN + ni * 16) * LDA], LDA, lane);
    #pragma unroll
    for (int mi = 0; mi < 4; ++mi)
      #pragma unroll
      for (int ni = 0; ni < 2; ++ni)
        acc[mi][ni] = wmma_bf16(af[mi], bfr[ni], acc[mi][ni]);

    if (kt + 1 < DM / 32) {
      lstoreB(cur ^ 1);
      wait_async0();
      __syncthreads();
    }
  }

  const int cn = lane & 15;
  const int rb = (lane >> 4) * 8;
  #pragma unroll
  for (int mi = 0; mi < 4; ++mi)
    #pragma unroll
    for (int ni = 0; ni < 2; ++ni) {
      const int gn = colBase + waveN + ni * 16 + cn;
      const float bv_ = bo[gn];
      #pragma unroll
      for (int i = 0; i < 8; ++i) {
        const int gm = rowBase + waveM + mi * 16 + rb + i;
        out[(size_t)gm * DM + gn] = acc[mi][ni][i] + bv_;
      }
    }
}

// ---------------------------------------------------------------------------
extern "C" void kernel_launch(void* const* d_in, const int* in_sizes, int n_in,
                              void* d_out, int out_size, void* d_ws, size_t ws_size,
                              hipStream_t stream) {
  const float* x  = (const float*)d_in[0];
  const float* Wq = (const float*)d_in[1];
  const float* bq = (const float*)d_in[2];
  const float* Wk = (const float*)d_in[3];
  const float* bk = (const float*)d_in[4];
  const float* Wv = (const float*)d_in[5];
  const float* bv = (const float*)d_in[6];
  const float* Wo = (const float*)d_in[7];
  const float* bo = (const float*)d_in[8];
  float* out = (float*)d_out;

  const size_t N = (size_t)8192 * DM;           // 8,388,608 elements per tensor
  __bf16* Qws  = (__bf16*)d_ws;
  __bf16* Kws  = Qws  + N;
  __bf16* Vtws = Kws  + N;
  __bf16* AOws = Vtws + N;                      // total 64 MB bf16 workspace

  qkv_gemm_kernel<<<dim3(64, 8, 3), 256, 0, stream>>>(x, Wq, bq, Wk, bk, Wv, bv,
                                                      Qws, Kws, Vtws);
  flash_attn_kernel<<<dim3(SEQ / 128, NH, 4), 256, 0, stream>>>(Qws, Kws, Vtws, AOws);
  out_gemm_kernel<<<dim3(64, 8), 256, 0, stream>>>(AOws, Wo, bo, out);
}
